// Model_37529424232711
// MI455X (gfx1250) — matrix operand order repaired and checked by reading, not compiled, not run on hardware
//
#include <hip/hip_runtime.h>

#define K_NN   20
#define B_SZ   16
#define N_PTS  1024
#define CAT_D  512

typedef __attribute__((ext_vector_type(16))) _Float16 v16h;
typedef __attribute__((ext_vector_type(8)))  _Float16 v8h;
typedef __attribute__((ext_vector_type(8)))  float    v8f;
typedef __attribute__((ext_vector_type(2)))  float    v2f;
typedef __attribute__((ext_vector_type(4)))  unsigned int v4u;
typedef __attribute__((ext_vector_type(4)))  int v4i;
typedef __attribute__((ext_vector_type(8)))  int v8i;

#if defined(__has_builtin)
#if __has_builtin(__builtin_amdgcn_tensor_load_to_lds)
#define HAVE_TDM 1
#endif
#if __has_builtin(__builtin_amdgcn_wmma_f32_16x16x4_f32)
#define HAVE_WMMA_F32 1
#endif
#endif
#ifndef HAVE_TDM
#define HAVE_TDM 0
#endif
#ifndef HAVE_WMMA_F32
#define HAVE_WMMA_F32 0
#endif

union HV { v16h v; v8h h[2]; };

__device__ __forceinline__ v8f wmma16x16x32(v16h a, v16h b, v8f c) {
  return __builtin_amdgcn_wmma_f32_16x16x32_f16(false, a, false, b, (short)0, c,
                                                false, false);
}

// f16 A-fragment (16x32, MxK): lane l -> row M=l&15; group g=l>>4 holds
// K = {g*8+0..7} U {16+g*8+0..7}   (ISA 7.12.2)
__device__ __forceinline__ v16h load_a_frag(const _Float16* p, int g) {
  HV u;
  u.h[0] = *(const v8h*)(p + g * 8);
  u.h[1] = *(const v8h*)(p + 16 + g * 8);
  return u.v;
}
// f16 B-fragment (32x16, KxN): lane l -> column N=l&15; group g holds
// K = {g*8+0..7} U {16+g*8+0..7}
__device__ __forceinline__ v16h load_b_frag(const _Float16* p, int g) {
  HV u;
  u.h[0] = *(const v8h*)(p + g * 8);
  u.h[1] = *(const v8h*)(p + 16 + g * 8);
  return u.v;
}

#if HAVE_TDM
// TDM: DMA a [rows x elems_x] tile (element size 2^ds_code bytes, row stride
// stride_elems elements) from global into LDS at byte offset lds_off.
// D# per ISA ch.8: group0 = {count=1, lds_addr, ga[31:0], ga[56:32]|type2<<30}
__device__ __forceinline__ void tdm_load_tile_2d(unsigned int lds_off,
                                                 const void* gptr, int elems_x,
                                                 int rows, int stride_elems,
                                                 int ds_code) {
  unsigned long long ga = (unsigned long long)(uintptr_t)gptr;
  v4u g0;
  g0[0] = 1u;                                   // count=1, user descriptor
  g0[1] = lds_off;                              // lds_addr (bytes)
  g0[2] = (unsigned int)ga;                     // global_addr[31:0]
  g0[3] = (unsigned int)((ga >> 32) & 0x01FFFFFFu) | (2u << 30);  // type=2
  v8i g1;
  g1[0] = (ds_code << 16);                      // data_size
  g1[1] = (elems_x & 0xFFFF) << 16;             // tensor_dim0[15:0]
  g1[2] = ((elems_x >> 16) & 0xFFFF) | ((rows & 0xFFFF) << 16);  // dim0hi|dim1
  g1[3] = (elems_x & 0xFFFF) << 16;             // tile_dim0
  g1[4] = rows & 0xFFFF;                        // tile_dim1 (tile_dim2=0)
  g1[5] = stride_elems;                         // tensor_dim0_stride[31:0]
  g1[6] = 0;
  g1[7] = 0;
  v4i z4 = {0, 0, 0, 0};
#if __clang_major__ >= 23
  v8i z8 = {0, 0, 0, 0, 0, 0, 0, 0};
  __builtin_amdgcn_tensor_load_to_lds(g0, g1, z4, z4, z8, 0);
#else
  __builtin_amdgcn_tensor_load_to_lds(g0, g1, z4, z4, 0);
#endif
}
#endif

// ---------------------------------------------------------------------------
// f16 WMMA NT-GEMM, TDM-staged LDS B panel:  C[m,n] = sum_k A[m,k]*Bw[n,k]
// Block = 8 waves sharing one 16-column B panel; one 16x16 tile per wave.
// ---------------------------------------------------------------------------
template <int KCH>
__global__ void wmma_gemm_nt_lds(const _Float16* __restrict__ A, int lda,
                                 long aBatch, const _Float16* __restrict__ Bw,
                                 int ldb, long bBatch, float* __restrict__ C,
                                 int ldc, long cBatch) {
  __shared__ _Float16 bpanel[16 * KCH * 32];
  const int lane = threadIdx.x & 31;
  const int wave = threadIdx.x >> 5;
  const int g = lane >> 4;
  const int r = lane & 15;
  const int n0 = blockIdx.y * 16;
  const _Float16* bsrc = Bw + (long)blockIdx.z * bBatch + (long)n0 * ldb;

#if HAVE_TDM
  if (threadIdx.x < 32) {  // one TDM op per workgroup (EXEC ignored by TDM)
    tdm_load_tile_2d((unsigned int)(uintptr_t)(void*)bpanel, bsrc, KCH * 32, 16,
                     ldb, 1);
    __builtin_amdgcn_s_wait_tensorcnt(0);
  }
#else
  for (int t = threadIdx.x; t < 16 * KCH * 32; t += 256) {
    int row = t / (KCH * 32), col = t % (KCH * 32);
    bpanel[t] = bsrc[(long)row * ldb + col];
  }
#endif
  __syncthreads();

  const int m0 = (blockIdx.x * 8 + wave) * 16;
  const _Float16* a = A + (long)blockIdx.z * aBatch + (long)(m0 + r) * lda;
  float* c = C + (long)blockIdx.z * cBatch;

  v8f acc = {};
#pragma unroll
  for (int kc = 0; kc < KCH; ++kc) {
    v16h bf = load_b_frag(&bpanel[r * (KCH * 32) + kc * 32], g);  // ds_load
    acc = wmma16x16x32(load_a_frag(a + kc * 32, g), bf, acc);
  }
#pragma unroll
  for (int i = 0; i < 8; ++i)
    c[(long)(m0 + i + 8 * g) * ldc + n0 + r] = acc[i];
}

#if HAVE_WMMA_F32
// ---------------------------------------------------------------------------
// fp32 WMMA Gram (exact distances for kNN): V_WMMA_F32_16X16X4_F32.
// A 16x4 layout (ISA 7.12.2): lane l -> row M=l&15; group g holds K=g*2+{0,1}
// in 2 VGPRs. B 4x16 symmetric: column N=l&15, K=g*2+{0,1}.
// ---------------------------------------------------------------------------
template <int KC4>  // number of K=4 chunks
__global__ void gram_f32_wmma(const float* __restrict__ A, int lda, long aBatch,
                              const float* __restrict__ Bw, int ldb,
                              long bBatch, float* __restrict__ C, int ldc,
                              long cBatch) {
  __shared__ float bpanel[16 * KC4 * 4];
  const int lane = threadIdx.x & 31;
  const int wave = threadIdx.x >> 5;
  const int g = lane >> 4;
  const int r = lane & 15;
  const int n0 = blockIdx.y * 16;
  const float* bsrc = Bw + (long)blockIdx.z * bBatch + (long)n0 * ldb;

#if HAVE_TDM
  if (threadIdx.x < 32) {
    tdm_load_tile_2d((unsigned int)(uintptr_t)(void*)bpanel, bsrc, KC4 * 4, 16,
                     ldb, 2);
    __builtin_amdgcn_s_wait_tensorcnt(0);
  }
#else
  for (int t = threadIdx.x; t < 16 * KC4 * 4; t += 256) {
    int row = t / (KC4 * 4), col = t % (KC4 * 4);
    bpanel[t] = bsrc[(long)row * ldb + col];
  }
#endif
  __syncthreads();

  const int m0 = (blockIdx.x * 8 + wave) * 16;
  const float* a = A + (long)blockIdx.z * aBatch + (long)(m0 + r) * lda;
  float* c = C + (long)blockIdx.z * cBatch;

  v8f acc = {};
#pragma unroll
  for (int kc = 0; kc < KC4; ++kc) {
    v2f af = *(const v2f*)(a + kc * 4 + g * 2);
    v2f bf = *(const v2f*)(&bpanel[r * (KC4 * 4) + kc * 4 + g * 2]);  // ds_load
    acc = __builtin_amdgcn_wmma_f32_16x16x4_f32(false, af, false, bf, (short)0,
                                                acc, false, false);
  }
#pragma unroll
  for (int i = 0; i < 8; ++i)
    c[(long)(m0 + i + 8 * g) * ldc + n0 + r] = acc[i];
}
#endif  // HAVE_WMMA_F32

// ---------------------------------------------------------------------------
// Fused EdgeConv: out = leaky( max_k WMMA(xj_k - xi, theta^T) + WMMA(xi,phi^T) )
// theta/phi/xi fragments register-resident across all 20 neighbors.
// Writes f16 into concat slice and fp32 into the fp32 activation buffer
// (consumed by the exact-fp32 Gram of the next layer).
// ---------------------------------------------------------------------------
template <int KCH>
__global__ void edge_conv_wmma(const _Float16* __restrict__ Hin, int ldh,
                               const _Float16* __restrict__ Th,  // [Dout,K]
                               const _Float16* __restrict__ Ph,  // [Dout,K]
                               const int* __restrict__ knn,      // [B*N,K_NN]
                               _Float16* __restrict__ Hcat,
                               float* __restrict__ Hf32, int ldcat,
                               int cat_off) {
  const int lane = threadIdx.x;
  const int g = lane >> 4;
  const int r = lane & 15;
  const int p = blockIdx.x * 16 + r;  // flattened b*N + n (A row)
  const int b = p >> 10;
  const int o0 = blockIdx.y * 16;
  const _Float16* xi = Hin + (long)p * ldh;
  const _Float16* th = Th + (long)(o0 + r) * (KCH * 32);
  const _Float16* ph = Ph + (long)(o0 + r) * (KCH * 32);

  v16h bth[KCH];           // theta fragments cached in registers
  v8h xil[KCH], xih[KCH];  // xi A-fragment halves cached
  v8f cphi = {};
#pragma unroll
  for (int c = 0; c < KCH; ++c) {
    const _Float16* pk = xi + c * 32;
    xil[c] = *(const v8h*)(pk + g * 8);
    xih[c] = *(const v8h*)(pk + 16 + g * 8);
    bth[c] = load_b_frag(th + c * 32, g);
    HV u;
    u.h[0] = xil[c];
    u.h[1] = xih[c];
    cphi = wmma16x16x32(u.v, load_b_frag(ph + c * 32, g), cphi);
  }

  const int* nb = knn + (long)p * K_NN;
  v8f best = {};
  for (int k = 0; k < K_NN; ++k) {
    const _Float16* xj = Hin + ((long)(b << 10) + nb[k]) * ldh;
    if (k + 1 < K_NN)
      __builtin_prefetch(Hin + ((long)(b << 10) + nb[k + 1]) * ldh, 0, 3);
    v8f cc = {};
#pragma unroll
    for (int c = 0; c < KCH; ++c) {
      HV u;
      u.h[0] = *(const v8h*)(xj + c * 32 + g * 8) - xil[c];
      u.h[1] = *(const v8h*)(xj + c * 32 + 16 + g * 8) - xih[c];
      cc = wmma16x16x32(u.v, bth[c], cc);
    }
    if (k == 0) {
      best = cc;
    } else {
#pragma unroll
      for (int i = 0; i < 8; ++i) best[i] = fmaxf(best[i], cc[i]);
    }
  }
#pragma unroll
  for (int i = 0; i < 8; ++i) {
    float v = best[i] + cphi[i];
    v = (v > 0.f) ? v : 0.2f * v;  // leaky_relu 0.2
    const long row = (long)(blockIdx.x * 16 + i + 8 * g);
    Hcat[row * ldcat + cat_off + o0 + r] = (_Float16)v;
    Hf32[row * ldcat + cat_off + o0 + r] = v;
  }
}

// ---------------------------------------------------------------------------
// Top-20 NN: one wave per row; 20 wave-wide argmin rounds (shfl_xor
// butterfly), ties toward lower index (stable top_k).
// ---------------------------------------------------------------------------
__global__ void knn_topk(const float* __restrict__ G,
                         const float* __restrict__ sq, int* __restrict__ knn) {
  const int p = blockIdx.x;  // b*1024 + n
  const int b = p >> 10;
  const int n = p & 1023;
  const int lane = threadIdx.x;
  const float* grow = G + ((long)b * N_PTS + n) * N_PTS;
  const float* sqb = sq + b * N_PTS;
  const float sn = sq[p];
  float d[32];
#pragma unroll
  for (int c = 0; c < 32; ++c) {
    int j = c * 32 + lane;
    d[c] = sn + sqb[j] - 2.f * grow[j];
  }
  for (int t = 0; t < K_NN; ++t) {
    float bv = 3.4e38f;
    int bj = N_PTS;
#pragma unroll
    for (int c = 0; c < 32; ++c) {
      int j = c * 32 + lane;
      bool take = (d[c] < bv) || (d[c] == bv && j < bj);
      bv = take ? d[c] : bv;
      bj = take ? j : bj;
    }
    for (int s = 16; s > 0; s >>= 1) {
      float ov = __shfl_xor(bv, s, 32);
      int oj = __shfl_xor(bj, s, 32);
      bool take = (ov < bv) || (ov == bv && oj < bj);
      bv = take ? ov : bv;
      bj = take ? oj : bj;
    }
    if ((bj & 31) == lane) {
      int cc = bj >> 5;
#pragma unroll
      for (int c = 0; c < 32; ++c)
        if (c == cc) d[c] = 3.4e38f;
    }
    if (lane == 0) knn[(long)p * K_NN + t] = bj;
  }
}

// ---------------------------------------------------------------------------
__global__ void cvt_pad_kernel(const float* __restrict__ src,
                               _Float16* __restrict__ dst, int rows, int cols,
                               int ldd) {
  int t = blockIdx.x * blockDim.x + threadIdx.x;
  if (t >= rows * ldd) return;
  int r = t / ldd, c = t % ldd;
  dst[t] = (c < cols) ? (_Float16)src[(long)r * cols + c] : (_Float16)0.f;
}

__global__ void cvt_pad_f32_kernel(const float* __restrict__ src,
                                   float* __restrict__ dst, int rows, int cols,
                                   int ldd) {
  int t = blockIdx.x * blockDim.x + threadIdx.x;
  if (t >= rows * ldd) return;
  int r = t / ldd, c = t % ldd;
  dst[t] = (c < cols) ? src[(long)r * cols + c] : 0.f;
}

__global__ void sqnorm_f32_kernel(const float* __restrict__ H, int ldh, int D,
                                  float* __restrict__ sq, int total) {
  int p = blockIdx.x * blockDim.x + threadIdx.x;
  if (p >= total) return;
  const float* row = H + (long)p * ldh;
  float s = 0.f;
  for (int d = 0; d < D; ++d) s += row[d] * row[d];
  sq[p] = s;
}

__global__ void sqnorm_f16_kernel(const _Float16* __restrict__ H, int ldh,
                                  int D, float* __restrict__ sq, int total) {
  int p = blockIdx.x * blockDim.x + threadIdx.x;
  if (p >= total) return;
  const _Float16* row = H + (long)p * ldh;
  float s = 0.f;
  for (int d = 0; d < D; ++d) {
    float v = (float)row[d];
    s += v * v;
  }
  sq[p] = s;
}

__global__ void pool_kernel(const float* __restrict__ Y,
                            const float* __restrict__ bias,
                            float* __restrict__ out) {
  int t = blockIdx.x * blockDim.x + threadIdx.x;
  if (t >= B_SZ * 1024) return;
  int b = t >> 10, o = t & 1023;
  const float* col = Y + (long)b * N_PTS * 1024 + o;
  float mx = -3.4e38f, sm = 0.f;
  for (int n = 0; n < N_PTS; ++n) {
    float v = col[(long)n * 1024];
    mx = fmaxf(mx, v);
    sm += v;
  }
  out[b * 2048 + o] = mx + bias[o];
  out[b * 2048 + 1024 + o] = sm * (1.f / N_PTS) + bias[o];
}

__global__ void fc_kernel(const float* __restrict__ X,
                          const float* __restrict__ W,
                          const float* __restrict__ bias, float* __restrict__ Y,
                          int M, int N, int Kd) {
  int t = blockIdx.x * blockDim.x + threadIdx.x;
  if (t >= M * N) return;
  int m = t / N, n = t % N;
  const float* x = X + (long)m * Kd;
  const float* w = W + (long)n * Kd;
  float acc = bias[n];
  for (int k = 0; k < Kd; ++k) acc += x[k] * w[k];
  Y[t] = acc;
}

__global__ void bn_leaky_kernel(float* __restrict__ Y,
                                const float* __restrict__ g,
                                const float* __restrict__ bvec, int M, int N) {
  int n = blockIdx.x * blockDim.x + threadIdx.x;
  if (n >= N) return;
  float mu = 0.f;
  for (int m = 0; m < M; ++m) mu += Y[m * N + n];
  mu /= M;
  float var = 0.f;
  for (int m = 0; m < M; ++m) {
    float d = Y[m * N + n] - mu;
    var += d * d;
  }
  var /= M;
  float s = rsqrtf(var + 1e-5f) * g[n];
  for (int m = 0; m < M; ++m) {
    float v = (Y[m * N + n] - mu) * s + bvec[n];
    Y[m * N + n] = (v > 0.f) ? v : 0.2f * v;
  }
}

// ---------------------------------------------------------------------------
extern "C" void kernel_launch(void* const* d_in, const int* in_sizes, int n_in,
                              void* d_out, int out_size, void* d_ws,
                              size_t ws_size, hipStream_t stream) {
  (void)in_sizes; (void)n_in; (void)out_size; (void)ws_size;
  const float* x = (const float*)d_in[0];
  const float* theta[4] = {(const float*)d_in[1], (const float*)d_in[3],
                           (const float*)d_in[5], (const float*)d_in[7]};
  const float* phi[4] = {(const float*)d_in[2], (const float*)d_in[4],
                         (const float*)d_in[6], (const float*)d_in[8]};
  const float* proj_w = (const float*)d_in[9];
  const float* proj_b = (const float*)d_in[10];
  const float* emb0_w = (const float*)d_in[11];
  const float* emb0_b = (const float*)d_in[12];
  const float* bn0_g = (const float*)d_in[13];
  const float* bn0_b = (const float*)d_in[14];
  const float* emb1_w = (const float*)d_in[15];
  const float* emb1_b = (const float*)d_in[16];
  const float* bn1_g = (const float*)d_in[17];
  const float* bn1_b = (const float*)d_in[18];
  const float* out_w = (const float*)d_in[19];
  const float* out_b = (const float*)d_in[20];

  const int P = B_SZ * N_PTS;  // 16384
  size_t off = 0;
  auto alloc = [&](size_t bytes) -> void* {
    void* p = (char*)d_ws + off;
    off += (bytes + 255) & ~(size_t)255;
    return p;
  };
  _Float16* hh_in = (_Float16*)alloc((size_t)P * 32 * 2);     // f16, pad 3->32
  _Float16* hh_cat = (_Float16*)alloc((size_t)P * CAT_D * 2); // f16 concat
  float* xpad = (float*)alloc((size_t)P * 4 * 4);             // fp32, pad 3->4
  float* hf32 = (float*)alloc((size_t)P * CAT_D * 4);         // fp32 acts
  float* gram = (float*)alloc((size_t)B_SZ * N_PTS * N_PTS * 4);  // also projY
  float* sq = (float*)alloc((size_t)P * 4);
  int* knn = (int*)alloc((size_t)P * K_NN * 4);
  float* pooled = (float*)alloc((size_t)B_SZ * 2048 * 4);
  float* t0 = (float*)alloc((size_t)B_SZ * 512 * 4);
  float* t1 = (float*)alloc((size_t)B_SZ * 256 * 4);

  const int DinA[4] = {3, 64, 64, 128};
  const int Kpad[4] = {32, 64, 64, 128};
  const int Dout[4] = {64, 64, 128, 256};
  const int catoff[4] = {0, 64, 128, 256};
  _Float16 *th16[4], *ph16[4];
  for (int i = 0; i < 4; ++i) {
    th16[i] = (_Float16*)alloc((size_t)Dout[i] * Kpad[i] * 2);
    ph16[i] = (_Float16*)alloc((size_t)Dout[i] * Kpad[i] * 2);
  }
  _Float16* pw16 = (_Float16*)alloc((size_t)1024 * 512 * 2);

  auto cdiv = [](int a, int b) { return (a + b - 1) / b; };

  cvt_pad_kernel<<<cdiv(P * 32, 256), 256, 0, stream>>>(x, hh_in, P, 3, 32);
  cvt_pad_f32_kernel<<<cdiv(P * 4, 256), 256, 0, stream>>>(x, xpad, P, 3, 4);
  for (int i = 0; i < 4; ++i) {
    cvt_pad_kernel<<<cdiv(Dout[i] * Kpad[i], 256), 256, 0, stream>>>(
        theta[i], th16[i], Dout[i], DinA[i], Kpad[i]);
    cvt_pad_kernel<<<cdiv(Dout[i] * Kpad[i], 256), 256, 0, stream>>>(
        phi[i], ph16[i], Dout[i], DinA[i], Kpad[i]);
  }
  cvt_pad_kernel<<<cdiv(1024 * 512, 256), 256, 0, stream>>>(proj_w, pw16, 1024,
                                                            512, 512);

  // 4 EdgeConv layers with dynamic kNN
  for (int i = 0; i < 4; ++i) {
    const _Float16* hin = (i == 0) ? hh_in : (hh_cat + catoff[i - 1]);
    const int ldh = (i == 0) ? 32 : CAT_D;
    dim3 gg(N_PTS / (16 * 8), N_PTS / 16, B_SZ);
#if HAVE_WMMA_F32
    // exact fp32 distances: Gram on the matrix pipe via WMMA_F32_16X16X4_F32
    const float* gin = (i == 0) ? xpad : (hf32 + catoff[i - 1]);
    const int ldg = (i == 0) ? 4 : CAT_D;
    const int kc4 = ((i == 0) ? 4 : DinA[i]) / 4;  // 1,16,16,32 chunks
    sqnorm_f32_kernel<<<cdiv(P, 256), 256, 0, stream>>>(gin, ldg, kc4 * 4, sq,
                                                        P);
    const long gab = (long)N_PTS * ldg;
    const long gcb = (long)N_PTS * N_PTS;
    switch (kc4) {
      case 1:
        gram_f32_wmma<1><<<gg, 256, 0, stream>>>(gin, ldg, gab, gin, ldg, gab,
                                                 gram, N_PTS, gcb);
        break;
      case 16:
        gram_f32_wmma<16><<<gg, 256, 0, stream>>>(gin, ldg, gab, gin, ldg, gab,
                                                  gram, N_PTS, gcb);
        break;
      default:
        gram_f32_wmma<32><<<gg, 256, 0, stream>>>(gin, ldg, gab, gin, ldg, gab,
                                                  gram, N_PTS, gcb);
        break;
    }
#else
    sqnorm_f16_kernel<<<cdiv(P, 256), 256, 0, stream>>>(hin, ldh, Kpad[i], sq,
                                                        P);
    const long hb = (long)N_PTS * ldh;
    const long gb = (long)N_PTS * N_PTS;
    switch (Kpad[i] / 32) {
      case 1:
        wmma_gemm_nt_lds<1><<<gg, 256, 0, stream>>>(hin, ldh, hb, hin, ldh, hb,
                                                    gram, N_PTS, gb);
        break;
      case 2:
        wmma_gemm_nt_lds<2><<<gg, 256, 0, stream>>>(hin, ldh, hb, hin, ldh, hb,
                                                    gram, N_PTS, gb);
        break;
      default:
        wmma_gemm_nt_lds<4><<<gg, 256, 0, stream>>>(hin, ldh, hb, hin, ldh, hb,
                                                    gram, N_PTS, gb);
        break;
    }
#endif
    knn_topk<<<P, 32, 0, stream>>>(gram, sq, knn);
    dim3 ge(P / 16, Dout[i] / 16, 1);
    switch (Kpad[i] / 32) {
      case 1:
        edge_conv_wmma<1><<<ge, 32, 0, stream>>>(hin, ldh, th16[i], ph16[i],
                                                 knn, hh_cat, hf32, CAT_D,
                                                 catoff[i]);
        break;
      case 2:
        edge_conv_wmma<2><<<ge, 32, 0, stream>>>(hin, ldh, th16[i], ph16[i],
                                                 knn, hh_cat, hf32, CAT_D,
                                                 catoff[i]);
        break;
      default:
        edge_conv_wmma<4><<<ge, 32, 0, stream>>>(hin, ldh, th16[i], ph16[i],
                                                 knn, hh_cat, hf32, CAT_D,
                                                 catoff[i]);
        break;
    }
  }

  // projection [P,512] x [1024,512]^T -> [P,1024] (reuses gram buffer)
  {
    dim3 gp(P / (16 * 8), 1024 / 16, 1);
    wmma_gemm_nt_lds<16><<<gp, 256, 0, stream>>>(hh_cat, CAT_D, 0, pw16, 512, 0,
                                                 gram, 1024, 0);
  }
  pool_kernel<<<cdiv(P, 256), 256, 0, stream>>>(gram, proj_b, pooled);

  // MLP head (fp32; dropout = identity)
  fc_kernel<<<cdiv(16 * 512, 256), 256, 0, stream>>>(pooled, emb0_w, emb0_b, t0,
                                                     16, 512, 2048);
  bn_leaky_kernel<<<cdiv(512, 128), 128, 0, stream>>>(t0, bn0_g, bn0_b, 16,
                                                      512);
  fc_kernel<<<cdiv(16 * 256, 256), 256, 0, stream>>>(t0, emb1_w, emb1_b, t1, 16,
                                                     256, 512);
  bn_leaky_kernel<<<cdiv(256, 128), 128, 0, stream>>>(t1, bn1_g, bn1_b, 16,
                                                      256);
  fc_kernel<<<cdiv(16 * 40, 256), 256, 0, stream>>>(t1, out_w, out_b,
                                                    (float*)d_out, 16, 40, 256);
}
